// LinearPolynomialCell_62285615727029
// MI455X (gfx1250) — matrix operand order repaired and checked by reading, not compiled, not run on hardware
//
#include <hip/hip_runtime.h>
#include <hip/hip_bf16.h>

// ---------------------------------------------------------------------------
// Problem constants (match reference)
// ---------------------------------------------------------------------------
static constexpr int DIM = 1024;
static constexpr int NB  = 16;           // batch
static constexpr int T   = 1024;         // timesteps
static constexpr int BD  = NB * DIM;     // 16384 elements per timestep
static constexpr int M   = T * NB;       // 16384 GEMM rows

typedef __bf16 bf16_t;
typedef bf16_t v16bf __attribute__((ext_vector_type(16)));
typedef bf16_t v8bf  __attribute__((ext_vector_type(8)));
typedef float  v8f   __attribute__((ext_vector_type(8)));

// LDS B-tile geometry: 128 rows (output columns) x 64 k per staged chunk.
// Row stride padded to 72 bf16 (144 B): 16B-aligned and conflict-free for the
// 16-lane ds_load_b128 fragment reads (stride 36 banks, all distinct mod 64).
static constexpr int LROW  = 72;              // bf16 elements per LDS row
static constexpr int LROWS = 128;             // rows per buffer
static constexpr int LBUF  = LROWS * LROW;    // elements per buffer
static constexpr int LBUFB = LBUF * 2;        // bytes per buffer
static constexpr int KC    = 64;              // staged k per chunk

__device__ __forceinline__ v16bf load_frag16(const bf16_t* p) {
    union { v16bf v; v8bf h[2]; } f;
    f.h[0] = *(const v8bf*)(p);
    f.h[1] = *(const v8bf*)(p + 16);
    return f.v;
}
__device__ __forceinline__ v16bf load_afrag(const bf16_t* p) {
    union { v16bf v; v8bf h[2]; } f;
    f.h[0] = *(const v8bf*)(p);
    f.h[1] = *(const v8bf*)(p + 16);
    return f.v;
}

// ---------------------------------------------------------------------------
// f32 -> bf16 conversion (grid-stride)
// ---------------------------------------------------------------------------
__global__ __launch_bounds__(256)
void cvt_f32_bf16(const float* __restrict__ in, bf16_t* __restrict__ out, int n) {
    int i = blockIdx.x * blockDim.x + threadIdx.x;
    int stride = gridDim.x * blockDim.x;
    for (; i < n; i += stride) out[i] = (bf16_t)in[i];
}

// ---------------------------------------------------------------------------
// bf16 GEMM with f32 accumulate: Dst[m,n] = epilogue( sum_k A[m,k]*W[n,k] )
//  MODE 0: z + bias            MODE 1: 1 + softplus(z + bias)
//  MODE 2: sigmoid(z + bias)   MODE 3: compete[m,n] * silu(z)
// Block = 256 threads (8 waves). Wave tile = 16 x 128. Block tile = 128 x 128.
// B tile staged in LDS (64 k per chunk) via async global->LDS copies,
// double-buffered over K; A fragments live in named VGPR double buffers.
// ---------------------------------------------------------------------------
template <int MODE>
__global__ __launch_bounds__(256)
void gemm_bf16_wmma(const bf16_t* __restrict__ A,        // [M,K] row-major
                    const bf16_t* __restrict__ W,        // [N,K] row-major
                    const float*  __restrict__ bias,     // [N] (MODE 0..2)
                    const float*  __restrict__ compete,  // [M,N] (MODE 3)
                    float* __restrict__ Dst,             // [M,N]
                    int Ndim, int K) {
    __shared__ bf16_t smem[2 * LBUF];

    const int tid  = threadIdx.x;
    const int lane = tid & 31;
    const int lm   = lane & 15;
    const int wave = tid >> 5;
    const int mBase = blockIdx.x * 128 + wave * 16;
    const int nBase = blockIdx.y * 128;

    // A fragment: lane<16 -> row mBase+lm, K {k0+0..7, k0+16..23}
    //             lane>=16 -> row mBase+lm, K {k0+8..15, k0+24..31}
    const int koffA = (lane >> 4) * 8;
    const bf16_t* __restrict__ arow = A + (size_t)(mBase + lm) * K + koffA;

    // B fragment (LDS): lane holds column i*16+lm, k {koffB+0..7, koffB+16..23}
    const int koffB = (lane >> 4) * 8;

    // async fill: 256 threads copy 128 rows x 128 B -> 64 B (4 x b128) each
    const int fr   = tid >> 1;               // row 0..127
    const int half = tid & 1;                // 0 / 1 -> bytes 0..63 / 64..127
    const bf16_t* __restrict__ gsrc = W + (size_t)(nBase + fr) * K + half * 32;
    const unsigned ldsFill =
        (unsigned)(size_t)(&smem[0]) + (unsigned)fr * (LROW * 2) + (unsigned)half * 64;

    v8f acc[8] = {v8f{}, v8f{}, v8f{}, v8f{}, v8f{}, v8f{}, v8f{}, v8f{}};

    auto fill = [&](int c, int sel) {
        unsigned long long g = (unsigned long long)(const void*)(gsrc + (size_t)c * KC);
        unsigned l = ldsFill + (unsigned)sel * LBUFB;
        asm volatile("global_load_async_to_lds_b128 %0, %1, off"
                     :: "v"(l),       "v"(g)        : "memory");
        asm volatile("global_load_async_to_lds_b128 %0, %1, off"
                     :: "v"(l + 16u), "v"(g + 16ull) : "memory");
        asm volatile("global_load_async_to_lds_b128 %0, %1, off"
                     :: "v"(l + 32u), "v"(g + 32ull) : "memory");
        asm volatile("global_load_async_to_lds_b128 %0, %1, off"
                     :: "v"(l + 48u), "v"(g + 48ull) : "memory");
    };

    // 16 x 128 x 32 sub-step: ksub = 0 or 32 within the staged 64-k chunk
    auto mm = [&](const bf16_t* lbuf, v16bf av, int ksub) {
        const bf16_t* lb = lbuf + (size_t)lm * LROW + ksub + koffB;
        v16bf b0 = load_frag16(lb + 0 * 16 * LROW);
        v16bf b1 = load_frag16(lb + 1 * 16 * LROW);
        v16bf b2 = load_frag16(lb + 2 * 16 * LROW);
        v16bf b3 = load_frag16(lb + 3 * 16 * LROW);
        __builtin_amdgcn_sched_barrier(0);   // keep loads batched ahead of math
        acc[0] = __builtin_amdgcn_wmma_f32_16x16x32_bf16(false, av, false, b0, (short)0, acc[0], false, false);
        acc[1] = __builtin_amdgcn_wmma_f32_16x16x32_bf16(false, av, false, b1, (short)0, acc[1], false, false);
        acc[2] = __builtin_amdgcn_wmma_f32_16x16x32_bf16(false, av, false, b2, (short)0, acc[2], false, false);
        acc[3] = __builtin_amdgcn_wmma_f32_16x16x32_bf16(false, av, false, b3, (short)0, acc[3], false, false);
        v16bf b4 = load_frag16(lb + 4 * 16 * LROW);
        v16bf b5 = load_frag16(lb + 5 * 16 * LROW);
        v16bf b6 = load_frag16(lb + 6 * 16 * LROW);
        v16bf b7 = load_frag16(lb + 7 * 16 * LROW);
        __builtin_amdgcn_sched_barrier(0);
        acc[4] = __builtin_amdgcn_wmma_f32_16x16x32_bf16(false, av, false, b4, (short)0, acc[4], false, false);
        acc[5] = __builtin_amdgcn_wmma_f32_16x16x32_bf16(false, av, false, b5, (short)0, acc[5], false, false);
        acc[6] = __builtin_amdgcn_wmma_f32_16x16x32_bf16(false, av, false, b6, (short)0, acc[6], false, false);
        acc[7] = __builtin_amdgcn_wmma_f32_16x16x32_bf16(false, av, false, b7, (short)0, acc[7], false, false);
    };

    const int NC = K / KC;   // staged chunks; assumed even (K = 1024 -> 16)

    // prologue
    fill(0, 0);
    v16bf a0A = load_afrag(arow);
    v16bf a0B = load_afrag(arow + 32);
    v16bf a1A, a1B;

    // manually unrolled-by-2 pipeline: named fragments stay in VGPRs
    for (int c = 0; c + 2 <= NC; c += 2) {
        // --- chunk c (buffer 0) ---
        asm volatile("s_wait_asynccnt 0x0" ::: "memory");
        __syncthreads();                       // fills landed; buf1 readers done
        fill(c + 1, 1);
        a1A = load_afrag(arow + (size_t)(c + 1) * KC);
        a1B = load_afrag(arow + (size_t)(c + 1) * KC + 32);
        mm(smem, a0A, 0);
        mm(smem, a0B, 32);
        // --- chunk c+1 (buffer 1) ---
        asm volatile("s_wait_asynccnt 0x0" ::: "memory");
        __syncthreads();                       // fills landed; buf0 readers done
        if (c + 2 < NC) {
            fill(c + 2, 0);
            a0A = load_afrag(arow + (size_t)(c + 2) * KC);
            a0B = load_afrag(arow + (size_t)(c + 2) * KC + 32);
        }
        mm(smem + LBUF, a1A, 0);
        mm(smem + LBUF, a1B, 32);
    }

    // ---- epilogue ----
    // C/D layout: VGPR r, lanes 0-15 -> row mBase+r, lanes 16-31 -> row mBase+8+r
    const int ncol0 = nBase + lm;
    const int rbase = mBase + ((lane >> 4) << 3);
#pragma unroll
    for (int i = 0; i < 8; ++i) {
        const int col = ncol0 + i * 16;
        const float bv = (MODE == 3) ? 0.0f : bias[col];
#pragma unroll
        for (int r = 0; r < 8; ++r) {
            const size_t o = (size_t)(rbase + r) * Ndim + col;
            float z = acc[i][r] + bv;
            float y;
            if (MODE == 0) {
                y = z;
            } else if (MODE == 1) {            // 1 + softplus(z)
                y = 1.0f + ((z > 20.0f) ? z : __logf(1.0f + __expf(z)));
            } else if (MODE == 2) {            // sigmoid(z)
                y = 1.0f / (1.0f + __expf(-z));
            } else {                            // compete * silu(z)
                const float s = z / (1.0f + __expf(-z));
                y = compete[o] * s;
            }
            Dst[o] = y;
        }
    }
}

// ---------------------------------------------------------------------------
// Sequential scan: one thread per (b,d); group-of-32 softmax via wave32
// shuffles (GROUP_SIZE == 32 == wave width, groups aligned).
// ---------------------------------------------------------------------------
__global__ __launch_bounds__(256)
void scan_kernel(const float* __restrict__ vx,      // [T,BD]
                 const float* __restrict__ alpha,   // [T,BD]
                 const float* __restrict__ delta,   // [T,BD]
                 const float* __restrict__ h0,      // [BD]
                 const float* __restrict__ r_h,     // [DIM]
                 float*  __restrict__ h_out,        // [T+1,BD]  (d_out)
                 bf16_t* __restrict__ h_bf,         // [T,BD]    (ws)
                 float*  __restrict__ compete) {    // [T,BD]    (ws)
    const int idx = blockIdx.x * blockDim.x + threadIdx.x;   // [0, BD)
    const int d = idx & (DIM - 1);
    const float rh = r_h[d];
    float h = h0[idx];
    h_out[idx] = h;                                          // h[0] = h0

    for (int t = 0; t < T; ++t) {
        const size_t off = (size_t)t * BD + idx;
        const float v  = vx[off] + rh * h;
        const float a  = alpha[off];
        const float dl = delta[off];
        float av = fabsf(v);
        av = fminf(fmaxf(av, 1e-6f), 10.0f);
        const float cand = copysignf(__expf(a * __logf(av)), v);
        h = (1.0f - dl) * h + dl * cand;

        h_out[off + BD] = h;           // h[t+1]
        h_bf[off] = (bf16_t)h;

        float mx = h;
#pragma unroll
        for (int s = 16; s > 0; s >>= 1)
            mx = fmaxf(mx, __shfl_xor(mx, s, 32));
        const float e = __expf(h - mx);
        float sum = e;
#pragma unroll
        for (int s = 16; s > 0; s >>= 1)
            sum += __shfl_xor(sum, s, 32);
        compete[off] = e / sum;
    }
}

// ---------------------------------------------------------------------------
// Host-side orchestration
// ---------------------------------------------------------------------------
extern "C" void kernel_launch(void* const* d_in, const int* in_sizes, int n_in,
                              void* d_out, int out_size, void* d_ws, size_t ws_size,
                              hipStream_t stream) {
    const float* x       = (const float*)d_in[0];
    const float* h0      = (const float*)d_in[1];
    const float* W_x     = (const float*)d_in[2];
    const float* r_h     = (const float*)d_in[3];
    const float* b       = (const float*)d_in[4];
    const float* W_alpha = (const float*)d_in[5];
    const float* b_alpha = (const float*)d_in[6];
    const float* W_delta = (const float*)d_in[7];
    const float* b_delta = (const float*)d_in[8];
    const float* W_out   = (const float*)d_in[9];
    float* out = (float*)d_out;

    // Workspace layout
    char* ws = (char*)d_ws;
    size_t p = 0;
    bf16_t* x_bf  = (bf16_t*)(ws + p); p += (size_t)M   * DIM * sizeof(bf16_t); // 32 MB
    bf16_t* Wx_bf = (bf16_t*)(ws + p); p += (size_t)DIM * DIM * sizeof(bf16_t); //  2 MB
    bf16_t* Wa_bf = (bf16_t*)(ws + p); p += (size_t)DIM * DIM * sizeof(bf16_t);
    bf16_t* Wd_bf = (bf16_t*)(ws + p); p += (size_t)DIM * DIM * sizeof(bf16_t);
    bf16_t* Wo_bf = (bf16_t*)(ws + p); p += (size_t)DIM * DIM * sizeof(bf16_t);
    float*  vx    = (float*) (ws + p); p += (size_t)M * DIM * sizeof(float);    // 64 MB
    float*  alpha = (float*) (ws + p); p += (size_t)M * DIM * sizeof(float);
    float*  delta = (float*) (ws + p); p += (size_t)M * DIM * sizeof(float);
    bf16_t* h_bf  = (bf16_t*)(ws + p); p += (size_t)M * DIM * sizeof(bf16_t);   // 32 MB
    float*  comp  = (float*) (ws + p); p += (size_t)M * DIM * sizeof(float);    // 64 MB

    // 1) Precision conversion (one pass; GEMMs then stream bf16 only)
    cvt_f32_bf16<<<2048, 256, 0, stream>>>(x,       x_bf,  M * DIM);
    cvt_f32_bf16<<<256,  256, 0, stream>>>(W_x,     Wx_bf, DIM * DIM);
    cvt_f32_bf16<<<256,  256, 0, stream>>>(W_alpha, Wa_bf, DIM * DIM);
    cvt_f32_bf16<<<256,  256, 0, stream>>>(W_delta, Wd_bf, DIM * DIM);
    cvt_f32_bf16<<<256,  256, 0, stream>>>(W_out,   Wo_bf, DIM * DIM);

    // 2) Three batched x-projections (WMMA bf16 -> f32)
    dim3 gg(M / 128, DIM / 128);
    gemm_bf16_wmma<0><<<gg, 256, 0, stream>>>(x_bf, Wx_bf, b,       nullptr, vx,    DIM, DIM);
    gemm_bf16_wmma<1><<<gg, 256, 0, stream>>>(x_bf, Wa_bf, b_alpha, nullptr, alpha, DIM, DIM);
    gemm_bf16_wmma<2><<<gg, 256, 0, stream>>>(x_bf, Wd_bf, b_delta, nullptr, delta, DIM, DIM);

    // 3) Elementwise recurrence (independent per (b,d)); fills h region of d_out
    scan_kernel<<<BD / 256, 256, 0, stream>>>(vx, alpha, delta, h0, r_h,
                                              out, h_bf, comp);

    // 4) Output GEMM: outs = compete * silu(h[1..T] @ W_out^T)
    float* outs = out + (size_t)(T + 1) * BD;
    gemm_bf16_wmma<3><<<gg, 256, 0, stream>>>(h_bf, Wo_bf, nullptr, comp, outs, DIM, DIM);
}
